// Net_90907277787544
// MI455X (gfx1250) — matrix operand order repaired and checked by reading, not compiled, not run on hardware
//
#include <hip/hip_runtime.h>

#define B_ 1024
#define R_ 100

typedef __attribute__((ext_vector_type(16))) _Float16 v16h;
typedef __attribute__((ext_vector_type(8)))  _Float16 v8h;
typedef __attribute__((ext_vector_type(8)))  float    v8f;

union V8F { v8f v; float f[8]; };

#define CAT16(lo, hi) __builtin_shufflevector(lo, hi, \
    0,1,2,3,4,5,6,7,8,9,10,11,12,13,14,15)

// ---------------------------------------------------------------------------
// conv1: [B,1,98,98] -> SAME 3x3 -> relu -> maxpool2 -> [B,6,49,49] (f32)
// ---------------------------------------------------------------------------
__global__ void conv1_pool_kernel(const float* __restrict__ x,
                                  const float* __restrict__ w,
                                  const float* __restrict__ bias,
                                  float* __restrict__ out)
{
    int idx = blockIdx.x * blockDim.x + threadIdx.x;
    const int total = B_ * 6 * 49 * 49;
    if (idx >= total) return;
    int ox = idx % 49; int t = idx / 49;
    int oy = t % 49;   t /= 49;
    int c  = t % 6;    int b = t / 6;

    const float* xin = x + (long)b * 98 * 98;
    const float* wc  = w + c * 9;
    float bc = bias[c];

    float m = -3.4e38f;
    #pragma unroll
    for (int py = 0; py < 2; ++py)
    #pragma unroll
    for (int px = 0; px < 2; ++px) {
        int iy = 2 * oy + py;
        int ix = 2 * ox + px;
        float s = bc;
        #pragma unroll
        for (int ky = -1; ky <= 1; ++ky)
        #pragma unroll
        for (int kx = -1; kx <= 1; ++kx) {
            int yy = iy + ky, xx = ix + kx;
            if (yy >= 0 && yy < 98 && xx >= 0 && xx < 98)
                s += xin[yy * 98 + xx] * wc[(ky + 1) * 3 + (kx + 1)];
        }
        m = fmaxf(m, s);
    }
    out[idx] = fmaxf(m, 0.0f);
}

// ---------------------------------------------------------------------------
// conv2: [B,6,49,49] -> VALID 6x6 -> relu -> maxpool2 -> [B,4,22,22] (f32)
// ---------------------------------------------------------------------------
__global__ void conv2_pool_kernel(const float* __restrict__ in,
                                  const float* __restrict__ w,
                                  const float* __restrict__ bias,
                                  float* __restrict__ out)
{
    int idx = blockIdx.x * blockDim.x + threadIdx.x;
    const int total = B_ * 4 * 22 * 22;
    if (idx >= total) return;
    int ox = idx % 22; int t = idx / 22;
    int oy = t % 22;   t /= 22;
    int c  = t % 4;    int b = t / 4;

    const float* xin = in + (long)b * 6 * 49 * 49;
    float bc = bias[c];

    float m = -3.4e38f;
    #pragma unroll
    for (int py = 0; py < 2; ++py)
    #pragma unroll
    for (int px = 0; px < 2; ++px) {
        int iy = 2 * oy + py;
        int ix = 2 * ox + px;
        float s = bc;
        for (int ic = 0; ic < 6; ++ic) {
            const float* xp = xin + (long)ic * 49 * 49 + iy * 49 + ix;
            const float* wp = w + ((c * 6 + ic) * 36);
            #pragma unroll
            for (int ky = 0; ky < 6; ++ky)
            #pragma unroll
            for (int kx = 0; kx < 6; ++kx)
                s += xp[ky * 49 + kx] * wp[ky * 6 + kx];
        }
        m = fmaxf(m, s);
    }
    out[idx] = fmaxf(m, 0.0f);
}

// ---------------------------------------------------------------------------
// conv3: [B,4,22,22] -> VALID 3x3 -> relu -> maxpool2 -> flatten,
// written as zero-padded f16 [B, 224]
// ---------------------------------------------------------------------------
__global__ void conv3_pool_kernel(const float* __restrict__ in,
                                  const float* __restrict__ w,
                                  const float* __restrict__ bias,
                                  _Float16* __restrict__ out)
{
    int idx = blockIdx.x * blockDim.x + threadIdx.x;
    const int total = B_ * 224;
    if (idx >= total) return;
    int col = idx % 224;
    int b   = idx / 224;
    if (col >= 200) { out[idx] = (_Float16)0.0f; return; }

    int c  = col / 100;
    int r  = col % 100;
    int oy = r / 10;
    int ox = r % 10;

    const float* xin = in + (long)b * 4 * 22 * 22;
    float bc = bias[c];

    float m = -3.4e38f;
    #pragma unroll
    for (int py = 0; py < 2; ++py)
    #pragma unroll
    for (int px = 0; px < 2; ++px) {
        int iy = 2 * oy + py;
        int ix = 2 * ox + px;
        float s = bc;
        #pragma unroll
        for (int ic = 0; ic < 4; ++ic) {
            const float* xp = xin + (long)ic * 22 * 22 + iy * 22 + ix;
            const float* wp = w + ((c * 4 + ic) * 9);
            #pragma unroll
            for (int ky = 0; ky < 3; ++ky)
            #pragma unroll
            for (int kx = 0; kx < 3; ++kx)
                s += xp[ky * 22 + kx] * wp[ky * 3 + kx];
        }
        m = fmaxf(m, s);
    }
    out[idx] = (_Float16)fmaxf(m, 0.0f);
}

// ---------------------------------------------------------------------------
// Pack weights:  W[z][K][N] f32  ->  Wt[z][Np][Kp] f16 (transposed, 0-padded)
// ---------------------------------------------------------------------------
__global__ void pack_weights_kernel(const float* __restrict__ W,
                                    _Float16* __restrict__ Wt,
                                    int K, int N, int Kp, int Np,
                                    long inStride, long outStride, int total)
{
    int idx = blockIdx.x * blockDim.x + threadIdx.x;
    if (idx >= total) return;
    int k = idx % Kp; int t = idx / Kp;
    int n = t % Np;   int z = t / Np;
    float val = (k < K && n < N) ? W[(long)z * inStride + (long)k * N + n] : 0.0f;
    Wt[(long)z * outStride + (long)n * Kp + k] = (_Float16)val;
}

// ---------------------------------------------------------------------------
// WMMA GEMM on packed f16 operands, software-pipelined, branch-free body.
//   A : f16 [M rows, stride lda], K-padded (Kp multiple of 32, zeros in pad)
//   Wt: f16 per-head [Np][Kp]  (transposed weights)
//   C : padded f16 for next layer (F16OUT=1) or exact f32 (F16OUT=0)
// TILES = 16-col output tiles per wave (2 when Np%32==0, else 1).
// ---------------------------------------------------------------------------
template<int TILES, int F16OUT>
__global__ __launch_bounds__(128) void gemm_wmma_packed(
    const _Float16* __restrict__ Abase, const _Float16* __restrict__ Wtbase,
    const float* __restrict__ biasBase, void* __restrict__ Cbase,
    int Kp, int N, int Np, int lda, int ldc,
    long aStride, long wStride, long bStride, long cStride, int relu)
{
    const int lane   = threadIdx.x & 31;
    const int wave   = threadIdx.x >> 5;
    const int groups = (Np / 16 + TILES - 1) / TILES;
    const int grp    = blockIdx.x * 4 + wave;
    if (grp >= groups) return;                 // wave-uniform exit

    const int z = blockIdx.z;
    const _Float16* A  = Abase  + (long)z * aStride;
    const _Float16* Wt = Wtbase + (long)z * wStride;
    const float* bias  = biasBase + (long)z * bStride;

    const int mbase = blockIdx.y * 16;
    const int l15   = lane & 15;
    const int hi    = lane >> 4;
    const int col0  = grp * (TILES * 16) + l15;
    const int col1  = col0 + 16;

    // A (16-bit 16x32 layout): lane half holds K-chunks {hi*8..+7},{16+hi*8..+7}
    const _Float16* ap  = A  + (long)(mbase + l15) * lda + hi * 8;
    // B (16-bit 32x16 layout): lane half holds K-chunks {hi*8..+7},{16+hi*8..+7} of its column
    const _Float16* wp0 = Wt + (long)col0 * Kp + hi * 8;
    const _Float16* wp1 = Wt + (long)col1 * Kp + hi * 8;

    v8f acc0 = {}; v8f acc1 = {};
    const int steps = Kp >> 5;

    // ---- software pipeline: prologue loads ----
    v8h alo = *(const v8h*)(ap);
    v8h ahi = *(const v8h*)(ap + 16);
    v8h b0l = *(const v8h*)(wp0);
    v8h b0h = *(const v8h*)(wp0 + 16);
    v8h b1l, b1h;
    if (TILES == 2) {
        b1l = *(const v8h*)(wp1);
        b1h = *(const v8h*)(wp1 + 16);
    }

    for (int s = 1; s < steps; ++s) {
        const int k0 = s << 5;
        // issue next iteration's loads before consuming current registers
        v8h n_alo = *(const v8h*)(ap + k0);
        v8h n_ahi = *(const v8h*)(ap + k0 + 16);
        v8h n_b0l = *(const v8h*)(wp0 + k0);
        v8h n_b0h = *(const v8h*)(wp0 + k0 + 16);
        v8h n_b1l, n_b1h;
        if (TILES == 2) {
            n_b1l = *(const v8h*)(wp1 + k0);
            n_b1h = *(const v8h*)(wp1 + k0 + 16);
        }

        v16h av = CAT16(alo, ahi);
        acc0 = __builtin_amdgcn_wmma_f32_16x16x32_f16(
            false, av, false, CAT16(b0l, b0h), (short)0, acc0, false, false);
        if (TILES == 2)
            acc1 = __builtin_amdgcn_wmma_f32_16x16x32_f16(
                false, av, false, CAT16(b1l, b1h), (short)0, acc1, false, false);

        alo = n_alo; ahi = n_ahi;
        b0l = n_b0l; b0h = n_b0h;
        if (TILES == 2) { b1l = n_b1l; b1h = n_b1h; }
    }
    // ---- pipeline drain ----
    {
        v16h av = CAT16(alo, ahi);
        acc0 = __builtin_amdgcn_wmma_f32_16x16x32_f16(
            false, av, false, CAT16(b0l, b0h), (short)0, acc0, false, false);
        if (TILES == 2)
            acc1 = __builtin_amdgcn_wmma_f32_16x16x32_f16(
                false, av, false, CAT16(b1l, b1h), (short)0, acc1, false, false);
    }

    float bv0 = (col0 < N) ? bias[col0] : 0.0f;
    float bv1 = (TILES == 2 && col1 < N) ? bias[col1] : 0.0f;
    V8F r0; r0.v = acc0;
    V8F r1; r1.v = acc1;

    if (F16OUT) {
        _Float16* C = (_Float16*)Cbase + (long)z * cStride;
        #pragma unroll
        for (int v = 0; v < 8; ++v) {
            int m = mbase + v + hi * 8;        // C layout: VGPR v -> M=v (+8 hi)
            float v0 = r0.f[v] + bv0;
            if (relu) v0 = fmaxf(v0, 0.0f);
            if (col0 >= N) v0 = 0.0f;          // zero K-pad for next layer
            C[(long)m * ldc + col0] = (_Float16)v0;
            if (TILES == 2) {
                float v1 = r1.f[v] + bv1;
                if (relu) v1 = fmaxf(v1, 0.0f);
                if (col1 >= N) v1 = 0.0f;
                C[(long)m * ldc + col1] = (_Float16)v1;
            }
        }
    } else {
        float* C = (float*)Cbase + (long)z * cStride;
        #pragma unroll
        for (int v = 0; v < 8; ++v) {
            int m = mbase + v + hi * 8;
            if (col0 < N) {
                float v0 = r0.f[v] + bv0;
                if (relu) v0 = fmaxf(v0, 0.0f);
                C[(long)m * ldc + col0] = v0;
            }
            if (TILES == 2 && col1 < N) {
                float v1 = r1.f[v] + bv1;
                if (relu) v1 = fmaxf(v1, 0.0f);
                C[(long)m * ldc + col1] = v1;
            }
        }
    }
}

// ---------------------------------------------------------------------------
extern "C" void kernel_launch(void* const* d_in, const int* in_sizes, int n_in,
                              void* d_out, int out_size, void* d_ws, size_t ws_size,
                              hipStream_t stream)
{
    const float* x    = (const float*)d_in[0];
    const float* c1w  = (const float*)d_in[1];
    const float* c1b  = (const float*)d_in[2];
    const float* c2w  = (const float*)d_in[3];
    const float* c2b  = (const float*)d_in[4];
    const float* c3w  = (const float*)d_in[5];
    const float* c3b  = (const float*)d_in[6];
    const float* fc1w = (const float*)d_in[7];
    const float* fc1b = (const float*)d_in[8];
    const float* fc2w = (const float*)d_in[9];
    const float* fc2b = (const float*)d_in[10];
    const float* h1w  = (const float*)d_in[11];
    const float* h1b  = (const float*)d_in[12];
    const float* h2w  = (const float*)d_in[13];
    const float* h2b  = (const float*)d_in[14];
    const float* h3w  = (const float*)d_in[15];
    const float* h3b  = (const float*)d_in[16];
    const float* h4w  = (const float*)d_in[17];
    const float* h4b  = (const float*)d_in[18];
    float* out = (float*)d_out;

    // ---- byte arena over d_ws, 256B-aligned slices ----
    char* base = (char*)d_ws;
    size_t off = 0;
    auto carve = [&](size_t bytes) {
        char* p = base + off;
        off += (bytes + 255) & ~(size_t)255;
        return p;
    };
    float*    T1   = (float*)   carve((size_t)B_ * 6 * 49 * 49 * 4);  // 59.0 MB
    float*    T2   = (float*)   carve((size_t)B_ * 4 * 22 * 22 * 4);  //  7.9 MB
    _Float16* T3h  = (_Float16*)carve((size_t)B_ * 224 * 2);
    _Float16* F1h  = (_Float16*)carve((size_t)B_ * 416 * 2);
    _Float16* F2h  = (_Float16*)carve((size_t)B_ * 224 * 2);
    _Float16* H2h  = (_Float16*)carve((size_t)B_ * R_ * 96 * 2);      // 19.7 MB
    _Float16* H3h  = (_Float16*)carve((size_t)B_ * R_ * 64 * 2);      // 13.1 MB
    _Float16* fc1t = (_Float16*)carve((size_t)416 * 224 * 2);
    _Float16* fc2t = (_Float16*)carve((size_t)224 * 416 * 2);
    _Float16* h1t  = (_Float16*)carve((size_t)R_ * 160 * 224 * 2);    //  7.2 MB
    _Float16* h2t  = (_Float16*)carve((size_t)R_ * 96 * 160 * 2);
    _Float16* h3t  = (_Float16*)carve((size_t)R_ * 64 * 96 * 2);
    _Float16* h4t  = (_Float16*)carve((size_t)R_ * 16 * 64 * 2);
    _Float16* H1h  = (_Float16*)T1;   // reuse: T1 dead after conv2; 32.8 < 59 MB

    // ---- pack all weights into transposed padded f16 ----
    #define PACK(W, Wt, K, N, Kp, Np, heads)                                   \
    {                                                                          \
        int total = (heads) * (Np) * (Kp);                                     \
        pack_weights_kernel<<<(total + 255) / 256, 256, 0, stream>>>(          \
            W, Wt, K, N, Kp, Np, (long)(K) * (N), (long)(Np) * (Kp), total);   \
    }
    PACK(fc1w, fc1t, 200, 400, 224, 416, 1);
    PACK(fc2w, fc2t, 400, 200, 416, 224, 1);
    PACK(h1w,  h1t,  200, 140, 224, 160, R_);
    PACK(h2w,  h2t,  140,  84, 160,  96, R_);
    PACK(h3w,  h3t,   84,  42,  96,  64, R_);
    PACK(h4w,  h4t,   42,   4,  64,  16, R_);
    #undef PACK

    // ---- conv trunk ----
    {
        int total = B_ * 6 * 49 * 49;
        conv1_pool_kernel<<<(total + 255) / 256, 256, 0, stream>>>(x, c1w, c1b, T1);
    }
    {
        int total = B_ * 4 * 22 * 22;
        conv2_pool_kernel<<<(total + 255) / 256, 256, 0, stream>>>(T1, c2w, c2b, T2);
    }
    {
        int total = B_ * 224;
        conv3_pool_kernel<<<(total + 255) / 256, 256, 0, stream>>>(T2, c3w, c3b, T3h);
    }

    const int MT = B_ / 16;   // 64 M-tiles
    #define GX2(np) ((((np) / 32) + 3) / 4)    // wave-groups for TILES=2

    // fc1: [B,224p] x [224p,416p] -> relu -> F1h (f16, ld 416)
    gemm_wmma_packed<2, 1><<<dim3(GX2(416), MT, 1), 128, 0, stream>>>(
        T3h, fc1t, fc1b, F1h, 224, 400, 416, 224, 416,
        0L, 0L, 0L, 0L, 1);

    // fc2: -> F2h (f16, ld 224)
    gemm_wmma_packed<2, 1><<<dim3(GX2(224), MT, 1), 128, 0, stream>>>(
        F1h, fc2t, fc2b, F2h, 416, 200, 224, 416, 224,
        0L, 0L, 0L, 0L, 1);

    // h1: per r -> H1h [B, R, 160] (f16)
    gemm_wmma_packed<2, 1><<<dim3(GX2(160), MT, R_), 128, 0, stream>>>(
        F2h, h1t, h1b, H1h, 224, 140, 160, 224, R_ * 160,
        0L, (long)160 * 224, 140L, 160L, 1);

    // h2: per r -> H2h [B, R, 96] (f16)
    gemm_wmma_packed<2, 1><<<dim3(GX2(96), MT, R_), 128, 0, stream>>>(
        H1h, h2t, h2b, H2h, 160, 84, 96, R_ * 160, R_ * 96,
        160L, (long)96 * 160, 84L, 96L, 1);

    // h3: per r -> H3h [B, R, 64] (f16)
    gemm_wmma_packed<2, 1><<<dim3(GX2(64), MT, R_), 128, 0, stream>>>(
        H2h, h3t, h3b, H3h, 96, 42, 64, R_ * 96, R_ * 64,
        96L, (long)64 * 96, 42L, 64L, 1);

    // h4: per r -> d_out [B, 400] (f32, no relu), single 16-col tile
    gemm_wmma_packed<1, 0><<<dim3(1, MT, R_), 128, 0, stream>>>(
        H3h, h4t, h4b, out, 64, 4, 16, R_ * 64, R_ * 4,
        64L, (long)16 * 64, 4L, 4L, 0);

    #undef GX2
}
